// Qwen3Attention_85495618995064
// MI455X (gfx1250) — compile-verified, matrix operand order checked by reading, not run on hardware
//
#include <hip/hip_runtime.h>
#include <hip/hip_bf16.h>

// ---------------------------------------------------------------------------
// Qwen3 attention block on gfx1250 (MI455X), bf16 WMMA everywhere.
//   B=1, S=4096, HID=2048, H=16, KVH=8, D=128, causal.
// ---------------------------------------------------------------------------

#define S_LEN    4096
#define HID_DIM  2048
#define NH_Q     16
#define NH_KV    8
#define DH       128
#define ATT_SCALE 0.08838834764831845f   // 128^-0.5
#define NEG_BIG  (-1.0e30f)

typedef __bf16 bf16;
typedef __attribute__((ext_vector_type(8)))  __bf16 bf16x8;
typedef __attribute__((ext_vector_type(16))) __bf16 bf16x16;
typedef __attribute__((ext_vector_type(8)))  float  f32x8;

static __device__ inline f32x8 zero8() {
  f32x8 z;
#pragma unroll
  for (int i = 0; i < 8; ++i) z[i] = 0.0f;
  return z;
}

static __device__ inline f32x8 wmma_bf16(bf16x16 a, bf16x16 b, f32x8 c) {
  // D = A(16x32 bf16) * B(32x16 bf16) + C(16x16 f32)
  return __builtin_amdgcn_wmma_f32_16x16x32_bf16(
      /*neg_a=*/false, a, /*neg_b=*/false, b,
      /*c_mod=*/(short)0, c, /*reuse_a=*/false, /*reuse_b=*/false);
}

// Load one 16(row)x32(K) bf16 WMMA operand fragment from a row-major matrix.
// Per ISA 7.12.2 (16-bit A 16x32): lane L holds row (L&15); lanes 0-15 carry
// K = 0..7 and 16..23, lanes 16-31 carry K = 8..15 and 24..31 (two 16B loads).
// Works for both A operands and B operands (B^T rows == weight rows).
static __device__ inline bf16x16 load_frag(const bf16* __restrict__ base,
                                           int ld, int row0, int k0, int lane) {
  const int r  = row0 + (lane & 15);
  const int ks = (lane >> 4) << 3;                 // 0 or 8
  const bf16* p = base + (size_t)r * ld + (k0 + ks);
  bf16x8 lo = *(const bf16x8*)p;                   // K = ks+0..7
  bf16x8 hi = *(const bf16x8*)(p + 16);            // K = 16+ks+0..7
  bf16x16 f;
#pragma unroll
  for (int i = 0; i < 8; ++i) { f[i] = lo[i]; f[i + 8] = hi[i]; }
  return f;
}

// ---------------------------------------------------------------------------
// f32 -> bf16 staging
// ---------------------------------------------------------------------------
__global__ void cvt_kernel(bf16* __restrict__ dst, const float* __restrict__ src, int n) {
  for (int i = blockIdx.x * blockDim.x + threadIdx.x; i < n;
       i += gridDim.x * blockDim.x)
    dst[i] = (bf16)src[i];
}

// ---------------------------------------------------------------------------
// Projection GEMM: C(32x128 tile) = X(MxK) * W^T(KxN) + bias, fused epilogue.
//   mode 0: RMSNorm(D=128) + RoPE, write bf16 out[h][s][d]   (Q and K)
//   mode 1: plain, write bf16 transposed out[h][d][s]        (V)
// Grid: (M/32, NH). One head per blockIdx.y since tileN == DH == 128.
// 4 waves; wave w owns columns w*32..w*32+31 of the tile.
// ---------------------------------------------------------------------------
__global__ __launch_bounds__(128) void proj_kernel(
    const bf16* __restrict__ X,        // [M][K]
    const bf16* __restrict__ W,        // [NH*128][K]
    const float* __restrict__ bias,    // [NH*128]
    const float* __restrict__ nw,      // [128] rmsnorm weight (mode 0)
    const float* __restrict__ cosp,    // [S][128] (mode 0)
    const float* __restrict__ sinp,    // [S][128] (mode 0)
    bf16* __restrict__ out,
    int K, int mode) {
  __shared__ float tile[32][DH];       // 16 KB

  const int m0   = blockIdx.x * 32;
  const int h    = blockIdx.y;
  const int n0   = h * DH;
  const int tid  = threadIdx.x;
  const int wave = tid >> 5;
  const int lane = tid & 31;
  const int nc   = n0 + wave * 32;

  f32x8 acc[2][2];
#pragma unroll
  for (int i = 0; i < 2; ++i)
#pragma unroll
    for (int j = 0; j < 2; ++j) acc[i][j] = zero8();

  for (int k0 = 0; k0 < K; k0 += 32) {
    __builtin_prefetch(X + (size_t)(m0 + (lane & 15)) * K + k0 + 128, 0, 0);
    bf16x16 a0 = load_frag(X, K, m0,      k0, lane);
    bf16x16 a1 = load_frag(X, K, m0 + 16, k0, lane);
    bf16x16 b0 = load_frag(W, K, nc,      k0, lane);
    bf16x16 b1 = load_frag(W, K, nc + 16, k0, lane);
    acc[0][0] = wmma_bf16(a0, b0, acc[0][0]);
    acc[0][1] = wmma_bf16(a0, b1, acc[0][1]);
    acc[1][0] = wmma_bf16(a1, b0, acc[1][0]);
    acc[1][1] = wmma_bf16(a1, b1, acc[1][1]);
  }

  // C layout: VGPR j -> row (lane>>4)*8 + j, col lane&15 (per 16x16 tile).
#pragma unroll
  for (int mi = 0; mi < 2; ++mi)
#pragma unroll
    for (int ni = 0; ni < 2; ++ni)
#pragma unroll
      for (int j = 0; j < 8; ++j) {
        int mm = mi * 16 + (lane >> 4) * 8 + j;
        int nn = wave * 32 + ni * 16 + (lane & 15);
        tile[mm][nn] = acc[mi][ni][j] + bias[n0 + nn];
      }
  __syncthreads();

  if (mode == 0) {
    // RMSNorm over D=128 then RoPE; one thread per row.
    if (tid < 32) {
      const int s = m0 + tid;
      float ss = 0.0f;
#pragma unroll 8
      for (int d = 0; d < DH; ++d) { float v = tile[tid][d]; ss += v * v; }
      const float inv = rsqrtf(ss * (1.0f / DH) + 1e-6f);
      bf16* o = out + ((size_t)h * S_LEN + s) * DH;
      const float* cr = cosp + (size_t)s * DH;
      const float* sr = sinp + (size_t)s * DH;
#pragma unroll 4
      for (int d = 0; d < DH / 2; ++d) {
        float xa = tile[tid][d]          * inv * nw[d];
        float xb = tile[tid][d + DH / 2] * inv * nw[d + DH / 2];
        o[d]          = (bf16)(xa * cr[d]          - xb * sr[d]);
        o[d + DH / 2] = (bf16)(xb * cr[d + DH / 2] + xa * sr[d + DH / 2]);
      }
    }
  } else {
    // V: write transposed out[h][d][s] so attention's P*V B-operand is
    // contiguous along keys.
    const int r = tid >> 2;              // row 0..31
    const int s = m0 + r;
    bf16* o = out + (size_t)h * DH * S_LEN;
#pragma unroll 8
    for (int d = (tid & 3) * 32; d < (tid & 3) * 32 + 32; ++d)
      o[(size_t)d * S_LEN + s] = (bf16)tile[r][d];
  }
}

// ---------------------------------------------------------------------------
// Flash attention, causal, GQA (kv head = h>>1).
// Grid (S/32, H); 4 waves. Wave w: S quadrant (mw,nw), O slice cols w*32..+31.
// ---------------------------------------------------------------------------
__global__ __launch_bounds__(128) void attn_kernel(
    const bf16* __restrict__ Q,   // [H][S][D]
    const bf16* __restrict__ Kc,  // [KVH][S][D]
    const bf16* __restrict__ Vt,  // [KVH][D][S]
    bf16* __restrict__ O) {       // [S][H*D]
  __shared__ float Sld[32][32];
  __shared__ __align__(16) bf16 Pld[32][32];
  __shared__ float Ald[32];
  __shared__ float Lld[32];

  const int q0   = blockIdx.x * 32;
  const int h    = blockIdx.y;
  const int tid  = threadIdx.x;
  const int wave = tid >> 5;
  const int lane = tid & 31;
  const int mw   = (wave >> 1) * 16;   // S quadrant row
  const int nw   = (wave & 1) * 16;    // S quadrant col

  const bf16* Qh = Q  + (size_t)h        * S_LEN * DH;
  const bf16* Kh = Kc + (size_t)(h >> 1) * S_LEN * DH;
  const bf16* Vh = Vt + (size_t)(h >> 1) * DH * S_LEN;

  bf16x16 qf[4];
#pragma unroll
  for (int kc = 0; kc < 4; ++kc)
    qf[kc] = load_frag(Qh, DH, q0 + mw, kc * 32, lane);

  f32x8 o[2][2];
#pragma unroll
  for (int i = 0; i < 2; ++i)
#pragma unroll
    for (int j = 0; j < 2; ++j) o[i][j] = zero8();

  float m_i = NEG_BIG, l_i = 0.0f;     // per-row state (threads 0..31)

  const int ktiles = q0 / 32 + 1;
  for (int t = 0; t < ktiles; ++t) {
    const int k0 = t * 32;

    // ---- S quadrant = Q * K^T over D=128 (4 chained WMMAs) ----
    f32x8 sfr = zero8();
#pragma unroll
    for (int kc = 0; kc < 4; ++kc) {
      bf16x16 kb = load_frag(Kh, DH, k0 + nw, kc * 32, lane);
      sfr = wmma_bf16(qf[kc], kb, sfr);
    }
#pragma unroll
    for (int j = 0; j < 8; ++j)
      Sld[mw + (lane >> 4) * 8 + j][nw + (lane & 15)] = sfr[j];
    __syncthreads();

    // ---- online softmax (one thread per query row) ----
    if (tid < 32) {
      const int qrow = q0 + tid;
      float sv[32];
      float rmax = NEG_BIG;
#pragma unroll
      for (int j = 0; j < 32; ++j) {
        float s = (k0 + j <= qrow) ? Sld[tid][j] * ATT_SCALE : NEG_BIG;
        sv[j] = s;
        rmax = fmaxf(rmax, s);
      }
      const float mnew  = fmaxf(m_i, rmax);
      const float alpha = __expf(m_i - mnew);    // first tile: exp(-huge)=0
      float sum = 0.0f;
#pragma unroll
      for (int j = 0; j < 32; ++j) {
        float p = __expf(sv[j] - mnew);          // masked -> underflow to 0
        Pld[tid][j] = (bf16)p;
        sum += p;
      }
      l_i = alpha * l_i + sum;
      m_i = mnew;
      Ald[tid] = alpha;
    }
    __syncthreads();

    // ---- rescale O and accumulate P*V ----
#pragma unroll
    for (int mi = 0; mi < 2; ++mi) {
#pragma unroll
      for (int j = 0; j < 8; ++j) {
        float a = Ald[mi * 16 + (lane >> 4) * 8 + j];
#pragma unroll
        for (int ni = 0; ni < 2; ++ni) o[mi][ni][j] *= a;
      }
    }
    bf16x16 p0 = load_frag(&Pld[0][0], 32, 0,  0, lane);
    bf16x16 p1 = load_frag(&Pld[0][0], 32, 16, 0, lane);
    bf16x16 v0 = load_frag(Vh, S_LEN, wave * 32,      k0, lane);
    bf16x16 v1 = load_frag(Vh, S_LEN, wave * 32 + 16, k0, lane);
    o[0][0] = wmma_bf16(p0, v0, o[0][0]);
    o[0][1] = wmma_bf16(p0, v1, o[0][1]);
    o[1][0] = wmma_bf16(p1, v0, o[1][0]);
    o[1][1] = wmma_bf16(p1, v1, o[1][1]);
  }

  if (tid < 32) Lld[tid] = l_i;
  __syncthreads();

#pragma unroll
  for (int mi = 0; mi < 2; ++mi)
#pragma unroll
    for (int ni = 0; ni < 2; ++ni)
#pragma unroll
      for (int j = 0; j < 8; ++j) {
        int mm = mi * 16 + (lane >> 4) * 8 + j;
        int nn = wave * 32 + ni * 16 + (lane & 15);
        float val = o[mi][ni][j] / Lld[mm];
        O[(size_t)(q0 + mm) * HID_DIM + h * DH + nn] = (bf16)val;
      }
}

// ---------------------------------------------------------------------------
// Output projection: d_out(f32) = OB(bf16) @ Wo^T(bf16). Tile 32x128.
// ---------------------------------------------------------------------------
__global__ __launch_bounds__(128) void oproj_kernel(
    const bf16* __restrict__ A,   // [S][2048]
    const bf16* __restrict__ W,   // [2048][2048]
    float* __restrict__ out) {    // [S][2048]
  const int m0   = blockIdx.x * 32;
  const int n0   = blockIdx.y * 128;
  const int wave = threadIdx.x >> 5;
  const int lane = threadIdx.x & 31;
  const int nc   = n0 + wave * 32;

  f32x8 acc[2][2];
#pragma unroll
  for (int i = 0; i < 2; ++i)
#pragma unroll
    for (int j = 0; j < 2; ++j) acc[i][j] = zero8();

  for (int k0 = 0; k0 < HID_DIM; k0 += 32) {
    bf16x16 a0 = load_frag(A, HID_DIM, m0,      k0, lane);
    bf16x16 a1 = load_frag(A, HID_DIM, m0 + 16, k0, lane);
    bf16x16 b0 = load_frag(W, HID_DIM, nc,      k0, lane);
    bf16x16 b1 = load_frag(W, HID_DIM, nc + 16, k0, lane);
    acc[0][0] = wmma_bf16(a0, b0, acc[0][0]);
    acc[0][1] = wmma_bf16(a0, b1, acc[0][1]);
    acc[1][0] = wmma_bf16(a1, b0, acc[1][0]);
    acc[1][1] = wmma_bf16(a1, b1, acc[1][1]);
  }
#pragma unroll
  for (int mi = 0; mi < 2; ++mi)
#pragma unroll
    for (int ni = 0; ni < 2; ++ni)
#pragma unroll
      for (int j = 0; j < 8; ++j) {
        int mm = m0 + mi * 16 + (lane >> 4) * 8 + j;
        int nn = nc + ni * 16 + (lane & 15);
        out[(size_t)mm * HID_DIM + nn] = acc[mi][ni][j];
      }
}

// ---------------------------------------------------------------------------
extern "C" void kernel_launch(void* const* d_in, const int* in_sizes, int n_in,
                              void* d_out, int out_size, void* d_ws, size_t ws_size,
                              hipStream_t stream) {
  const float* x    = (const float*)d_in[0];
  const float* cosp = (const float*)d_in[1];
  const float* sinp = (const float*)d_in[2];
  // d_in[3] = mask: causal handled analytically
  const float* Wq   = (const float*)d_in[4];
  const float* bq   = (const float*)d_in[5];
  const float* Wk   = (const float*)d_in[6];
  const float* bk   = (const float*)d_in[7];
  const float* Wv   = (const float*)d_in[8];
  const float* bv   = (const float*)d_in[9];
  const float* Wo   = (const float*)d_in[10];
  const float* qnw  = (const float*)d_in[11];
  const float* knw  = (const float*)d_in[12];

  char* w = (char*)d_ws;
  bf16* XB  = (bf16*)w; w += (size_t)S_LEN * HID_DIM * 2;          // x bf16
  bf16* WQB = (bf16*)w; w += (size_t)HID_DIM * HID_DIM * 2;        // Wq bf16
  bf16* WKB = (bf16*)w; w += (size_t)(NH_KV * DH) * HID_DIM * 2;   // Wk bf16
  bf16* WVB = (bf16*)w; w += (size_t)(NH_KV * DH) * HID_DIM * 2;   // Wv bf16
  bf16* WOB = (bf16*)w; w += (size_t)HID_DIM * HID_DIM * 2;        // Wo bf16
  bf16* QB  = (bf16*)w; w += (size_t)NH_Q  * S_LEN * DH * 2;       // [h][s][d]
  bf16* KB  = (bf16*)w; w += (size_t)NH_KV * S_LEN * DH * 2;       // [h][s][d]
  bf16* VTB = (bf16*)w; w += (size_t)NH_KV * DH * S_LEN * 2;       // [h][d][s]
  bf16* OB  = (bf16*)w; w += (size_t)S_LEN * HID_DIM * 2;          // attn out

  cvt_kernel<<<2048, 256, 0, stream>>>(XB,  x,  S_LEN * HID_DIM);
  cvt_kernel<<<2048, 256, 0, stream>>>(WQB, Wq, HID_DIM * HID_DIM);
  cvt_kernel<<<1024, 256, 0, stream>>>(WKB, Wk, (NH_KV * DH) * HID_DIM);
  cvt_kernel<<<1024, 256, 0, stream>>>(WVB, Wv, (NH_KV * DH) * HID_DIM);
  cvt_kernel<<<2048, 256, 0, stream>>>(WOB, Wo, HID_DIM * HID_DIM);

  proj_kernel<<<dim3(S_LEN / 32, NH_Q),  128, 0, stream>>>(
      XB, WQB, bq, qnw, cosp, sinp, QB, HID_DIM, 0);
  proj_kernel<<<dim3(S_LEN / 32, NH_KV), 128, 0, stream>>>(
      XB, WKB, bk, knw, cosp, sinp, KB, HID_DIM, 0);
  proj_kernel<<<dim3(S_LEN / 32, NH_KV), 128, 0, stream>>>(
      XB, WVB, bv, nullptr, nullptr, nullptr, VTB, HID_DIM, 1);

  attn_kernel<<<dim3(S_LEN / 32, NH_Q), 128, 0, stream>>>(QB, KB, VTB, OB);

  oproj_kernel<<<dim3(S_LEN / 32, HID_DIM / 128), 128, 0, stream>>>(
      OB, WOB, (float*)d_out);
}
